// DilatedToothSegmentationNetwork_39444979647066
// MI455X (gfx1250) — matrix operand order repaired and checked by reading, not compiled, not run on hardware
//
#include <hip/hip_runtime.h>
#include <hip/hip_bf16.h>

// ---------------------------------------------------------------------------
// DGCNN-style tooth segmentation forward, MI455X (gfx1250, wave32).
// Dense layers: bf16 WMMA (v_wmma_f32_16x16x32_bf16), f32 accumulate.
// Full tiles staged by the Tensor Data Mover (tensor_load_to_lds +
// s_wait_tensorcnt); one shared fp32->bf16 packed-convert pass per tile so the
// WMMA inner loop is pure ds_load_2addr_b32 -> v_wmma.
// Neighbor selection: per-row LDS bitonic sort (one sort of cd serves knn +
// all three dilated selectors). Edge convs: one wave per point, lane=neighbor.
// ---------------------------------------------------------------------------

typedef __attribute__((ext_vector_type(16))) __bf16 v16bf;
typedef __attribute__((ext_vector_type(2)))  __bf16 v2bf;
typedef __attribute__((ext_vector_type(8)))  float  v8f;
typedef __attribute__((ext_vector_type(4)))  unsigned int uint4v;
typedef __attribute__((ext_vector_type(8)))  int int8v;
typedef __attribute__((ext_vector_type(4)))  int int4v;

union BF16Frag { unsigned int u[8]; v16bf v; };

// packed f32x2 -> bf16x2 via native clang conversions (RNE)
__device__ __forceinline__ unsigned pack2(float lo, float hi) {
    union { v2bf v; unsigned u; } cv;
    cv.v = (v2bf){(__bf16)lo, (__bf16)hi};
    return cv.u;
}

// ---------------- TDM helpers (gfx1250 Tensor Data Mover) ------------------
#if __has_builtin(__builtin_amdgcn_tensor_load_to_lds)
#define HAS_TDM 1
#else
#define HAS_TDM 0
#endif

typedef __attribute__((address_space(3))) void lds_void_t;
__device__ __forceinline__ unsigned lds_addr_of(void* p) {
    return (unsigned)(uintptr_t)(lds_void_t*)p;   // LDS byte offset
}

// 2-D tile load: tile_d0 contiguous elements (4B each) per row, tile_d1 rows,
// row stride stride0 elements; contiguous LDS destination (no padding).
__device__ __forceinline__ void tdm_load_2d(
    unsigned lds_byte, const float* gptr, unsigned tensor_d0, unsigned tensor_d1,
    unsigned tile_d0, unsigned tile_d1, unsigned stride0)
{
#if HAS_TDM
    unsigned long long ga = (unsigned long long)(uintptr_t)gptr;
    uint4v g0;
    g0.x = 1u;                                        // count=1 (valid user D#)
    g0.y = lds_byte;                                  // lds_addr
    g0.z = (unsigned)ga;                              // global_addr[31:0]
    g0.w = (unsigned)((ga >> 32) & 0x01FFFFFFull)     // global_addr[56:32]
         | (2u << 30);                                // type = 2 ("image")
    int8v g1;
    g1[0] = (int)(2u << 16);                          // data_size = 4 bytes
    g1[1] = (int)((tensor_d0 & 0xFFFFu) << 16);       // tensor_dim0[15:0] @48
    g1[2] = (int)(((tensor_d0 >> 16) & 0xFFFFu)       // tensor_dim0[31:16]
                | ((tensor_d1 & 0xFFFFu) << 16));     // tensor_dim1[15:0]
    g1[3] = (int)(((tensor_d1 >> 16) & 0xFFFFu)       // tensor_dim1[31:16]
                | (tile_d0 << 16));                   // tile_dim0
    g1[4] = (int)(tile_d1 & 0xFFFFu);                 // tile_dim1 (tile_dim2=0)
    g1[5] = (int)stride0;                             // tensor_dim0_stride[31:0]
    g1[6] = 0;                                        // stride0 hi / stride1 lo
    g1[7] = 0;                                        // stride1 hi
    int4v z4 = {0, 0, 0, 0};
#if defined(__clang_major__) && (__clang_major__ >= 23)
    int8v z8 = {0, 0, 0, 0, 0, 0, 0, 0};
    __builtin_amdgcn_tensor_load_to_lds(g0, g1, z4, z4, z8, 0);
#else
    __builtin_amdgcn_tensor_load_to_lds(g0, g1, z4, z4, 0);
#endif
#endif
}

__device__ __forceinline__ void tdm_wait() {
#if __has_builtin(__builtin_amdgcn_s_wait_tensorcnt)
    __builtin_amdgcn_s_wait_tensorcnt(0);
#endif
}

// ---------------------------------------------------------------------------
// Generic GEMM: out[M,N] = epilogue(A[M,K] @ W[K,N] + bias)
// epi: 0=bias, 1=bias+relu, 2=aux*sigmoid(v), 3=relu(v+aux)
// Block 256 thr = 8 waves; block tile 64x64; wave tile 32x16 (2 WMMA accums).
// ---------------------------------------------------------------------------
__global__ __launch_bounds__(256) void k_gemm(
    const float* __restrict__ A, const float* __restrict__ W,
    const float* __restrict__ bias, const float* __restrict__ aux,
    float* __restrict__ out, int M, int Kd, int Nd, int epi)
{
    __shared__ float As[64][32];            // fp32 A tile (TDM destination)
    __shared__ float Bs[32][64];            // fp32 W tile (TDM destination)
    __shared__ unsigned int Apair[64][17];  // [m][k/2] packed bf16 pairs
    __shared__ unsigned int Bpair[64][17];  // [n][k/2] packed bf16 pairs

    const int tid  = threadIdx.x;
    const int wave = tid >> 5, lane = tid & 31;
    const int wm = wave >> 2, wn = wave & 3;
    const int bm = blockIdx.x * 64;
    const int bn = blockIdx.y * 64;

    const unsigned ldsA = lds_addr_of(&As[0][0]);
    const unsigned ldsB = lds_addr_of(&Bs[0][0]);

    v8f c0 = {}, c1 = {};

    for (int k0 = 0; k0 < Kd; k0 += 32) {
        const bool full = HAS_TDM && (bm + 64 <= M) && (k0 + 32 <= Kd) && (bn + 64 <= Nd);
        if (full) {
            if (wave == 0) {
                // A tile: 64 rows x 32 contiguous K, row stride Kd
                tdm_load_2d(ldsA, A + (size_t)bm * Kd + k0,
                            (unsigned)(Kd - k0), (unsigned)(M - bm), 32u, 64u,
                            (unsigned)Kd);
                // W tile: 32 rows x 64 contiguous N, row stride Nd
                tdm_load_2d(ldsB, W + (size_t)k0 * Nd + bn,
                            (unsigned)(Nd - bn), (unsigned)(Kd - k0), 64u, 32u,
                            (unsigned)Nd);
                tdm_wait();
            }
        } else {
            // branchless guarded staging (clamped address + select-zero)
            for (int p = tid; p < 64 * 32; p += 256) {
                int m = p >> 5, kk = p & 31;
                int gm = bm + m, gk = k0 + kk;
                int cm = gm < M ? gm : M - 1;
                int ck = gk < Kd ? gk : Kd - 1;
                float f = A[(size_t)cm * Kd + ck];
                As[m][kk] = (gm < M && gk < Kd) ? f : 0.f;
            }
            for (int p = tid; p < 32 * 64; p += 256) {
                int k = p >> 6, n = p & 63;
                int gk = k0 + k, gn = bn + n;
                int ck = gk < Kd ? gk : Kd - 1;
                int cn = gn < Nd ? gn : Nd - 1;
                float f = W[(size_t)ck * Nd + cn];
                Bs[k][n] = (gk < Kd && gn < Nd) ? f : 0.f;
            }
        }
        // prefetch next A K-slab into caches
        if (k0 + 32 < Kd && (bm + (tid & 63)) < M)
            __builtin_prefetch(&A[(size_t)(bm + (tid & 63)) * Kd + k0 + 32], 0, 1);
        __syncthreads();

        // shared packed-convert pass: each tile element converted exactly once
        for (int p = tid; p < 64 * 16; p += 256) {
            int m = p >> 4, kp = p & 15;
            Apair[m][kp] = pack2(As[m][2 * kp], As[m][2 * kp + 1]);
        }
        for (int p = tid; p < 64 * 16; p += 256) {
            int kp = p >> 6, n = p & 63;
            Bpair[n][kp] = pack2(Bs[2 * kp][n], Bs[2 * kp + 1][n]);
        }
        __syncthreads();

        // B fragment (32x16 bf16): VGPR0-3: K={0..7|8..15}, VGPR4-7: K={16..23|24..31} by lane half
        BF16Frag bf;
        {
            int nl = wn * 16 + (lane & 15);
            int kb = (lane < 16) ? 0 : 8;
#pragma unroll
            for (int r = 0; r < 8; ++r)
                bf.u[r] = Bpair[nl][((r < 4) ? (kb + 2 * r) : (16 + kb + 2 * (r - 4))) >> 1];
        }
#pragma unroll
        for (int t = 0; t < 2; ++t) {
            // A fragment (16x32 bf16): lanes0-15 M, VGPR0-3: K={0..7|8..15},
            // VGPR4-7: K={16..23|24..31} by lane half.
            BF16Frag af;
            int m     = wm * 32 + t * 16 + (lane & 15);
            int khalf = (lane < 16) ? 0 : 8;
#pragma unroll
            for (int r = 0; r < 8; ++r) {
                int kk = (r < 4) ? (khalf + 2 * r) : (16 + khalf + 2 * (r - 4));
                af.u[r] = Apair[m][kk >> 1];
            }
            if (t == 0)
                c0 = __builtin_amdgcn_wmma_f32_16x16x32_bf16(false, af.v, false, bf.v,
                                                             (short)0, c0, false, false);
            else
                c1 = __builtin_amdgcn_wmma_f32_16x16x32_bf16(false, af.v, false, bf.v,
                                                             (short)0, c1, false, false);
        }
        __syncthreads();
    }

    // epilogue + store.  C/D layout: VGPR r -> M=r (lanes0-15) / r+8 (lanes16-31), N=lane%16
    const int ncol = bn + wn * 16 + (lane & 15);
    const float bv = (ncol < Nd) ? bias[ncol] : 0.f;
#pragma unroll
    for (int t = 0; t < 2; ++t) {
        v8f acc = t ? c1 : c0;
#pragma unroll
        for (int r = 0; r < 8; ++r) {
            int mrow = bm + wm * 32 + t * 16 + r + ((lane < 16) ? 0 : 8);
            if (mrow < M && ncol < Nd) {
                size_t o = (size_t)mrow * Nd + ncol;
                float v = acc[r] + bv;
                if (epi == 1)      v = fmaxf(v, 0.f);
                else if (epi == 2) v = aux[o] * (1.f / (1.f + __expf(-v)));
                else if (epi == 3) v = fmaxf(v + aux[o], 0.f);
                out[o] = v;
            }
        }
    }
}

// ---------------------------------------------------------------------------
// Per-row distance + full bitonic sort; emits up to 4 strided rank sets.
// One block per (b, i) row; 8192 (dist,idx) pairs in LDS (48 KB).
// ---------------------------------------------------------------------------
template <int C>
__global__ __launch_bounds__(256) void k_knn_sort(
    const float* __restrict__ feat, int rstride,
    int* o0, int s0, int* o1, int s1, int* o2, int s2, int* o3, int s3)
{
    constexpr int NP = 8192;
    __shared__ float          d[NP];
    __shared__ unsigned short id[NP];
    __shared__ float          fc[C];

    const int b = blockIdx.x >> 13;          // / 8192
    const int i = blockIdx.x & (NP - 1);
    const float* base = feat + (size_t)b * NP * rstride;

    if (threadIdx.x < C) fc[threadIdx.x] = base[(size_t)i * rstride + threadIdx.x];
    __syncthreads();

    for (int j = threadIdx.x; j < NP; j += 256) {
        float s = 0.f;
        for (int c = 0; c < C; ++c) {
            float df = base[(size_t)j * rstride + c] - fc[c];
            s += df * df;
        }
        d[j]  = s;
        id[j] = (unsigned short)j;
    }
    __syncthreads();

    for (int ksz = 2; ksz <= NP; ksz <<= 1) {
        for (int jj = ksz >> 1; jj > 0; jj >>= 1) {
            for (int t = threadIdx.x; t < NP; t += 256) {
                int ixj = t ^ jj;
                if (ixj > t) {
                    bool up = (t & ksz) == 0;
                    float a = d[t], bb = d[ixj];
                    if (up ? (a > bb) : (a < bb)) {
                        d[t] = bb; d[ixj] = a;
                        unsigned short ti = id[t]; id[t] = id[ixj]; id[ixj] = ti;
                    }
                }
            }
            __syncthreads();
        }
    }

    const size_t row = blockIdx.x;
    if (threadIdx.x < 32) {
        if (o0) o0[row * 32 + threadIdx.x] = id[threadIdx.x * s0];
        if (o1) o1[row * 32 + threadIdx.x] = id[threadIdx.x * s1];
        if (o2) o2[row * 32 + threadIdx.x] = id[threadIdx.x * s2];
        if (o3) o3[row * 32 + threadIdx.x] = id[threadIdx.x * s3];
    }
}

// ---------------------------------------------------------------------------
// EdgeConv: e=[nbr-ctr, ctr] -> relu(W1)->relu(W2)-> max over 32 neighbors.
// One wave per point, lane = neighbor (k == warpSize == 32).
// ---------------------------------------------------------------------------
template <int CIN, int H>
__global__ __launch_bounds__(128) void k_edge_conv(
    const float* __restrict__ x, const int* __restrict__ idx,
    const float* __restrict__ W1, const float* __restrict__ b1,
    const float* __restrict__ W2, const float* __restrict__ b2,
    float* __restrict__ out)
{
    constexpr int NP = 8192;
    constexpr int E  = 2 * CIN;
    __shared__ float sW1[E][H];
    __shared__ float sW2[H][H];
    __shared__ float sb1[H];
    __shared__ float sb2[H];

    for (int p = threadIdx.x; p < E * H; p += 128) sW1[p / H][p % H] = W1[p];
    for (int p = threadIdx.x; p < H * H; p += 128) sW2[p / H][p % H] = W2[p];
    for (int p = threadIdx.x; p < H; p += 128) { sb1[p] = b1[p]; sb2[p] = b2[p]; }
    __syncthreads();

    const int wave  = threadIdx.x >> 5, lane = threadIdx.x & 31;
    const int point = blockIdx.x * 4 + wave;       // [0, B*NP)
    const int b     = point >> 13;
    const int n     = point & (NP - 1);
    const float* xb = x + (size_t)b * NP * CIN;
    const int nb    = idx[(size_t)point * 32 + lane];

    float e[E];
    for (int c = 0; c < CIN; ++c) {
        float ctr = xb[(size_t)n  * CIN + c];
        float nbr = xb[(size_t)nb * CIN + c];
        e[c]       = nbr - ctr;
        e[CIN + c] = ctr;
    }
    float h1[H];
    for (int j = 0; j < H; ++j) {
        float s = sb1[j];
        for (int i2 = 0; i2 < E; ++i2) s += e[i2] * sW1[i2][j];
        h1[j] = fmaxf(s, 0.f);
    }
    for (int j = 0; j < H; ++j) {
        float s = sb2[j];
        for (int i2 = 0; i2 < H; ++i2) s += h1[i2] * sW2[i2][j];
        s = fmaxf(s, 0.f);
#pragma unroll
        for (int off = 16; off > 0; off >>= 1)
            s = fmaxf(s, __shfl_xor(s, off, 32));
        if (lane == 0) out[(size_t)point * H + j] = s;
    }
}

// --------------------------- small utility kernels -------------------------
__global__ void k_transpose_in(const float* __restrict__ data, float* __restrict__ xT) {
    const int NP = 8192;
    long i = blockIdx.x * (long)blockDim.x + threadIdx.x;
    if (i >= 2L * 6 * NP) return;
    int n = (int)(i % NP); int c = (int)((i / NP) % 6); int b = (int)(i / (6L * NP));
    xT[((long)b * NP + n) * 6 + c] = data[i];
}

__global__ void k_colmax(const float* __restrict__ x, float* __restrict__ g, int C) {
    const int NP = 8192;
    int b = blockIdx.y;
    int c = blockIdx.x * blockDim.x + threadIdx.x;
    if (c >= C) return;
    const float* p = x + (size_t)b * NP * C + c;
    float m = -3.4e38f;
    for (int n = 0; n < NP; ++n) m = fmaxf(m, p[(size_t)n * C]);
    g[(size_t)b * C + c] = m;
}

__global__ void k_apply_stn(const float* __restrict__ x, const float* __restrict__ t36,
                            float* __restrict__ y) {
    const int NP = 8192;
    int gid = blockIdx.x * blockDim.x + threadIdx.x;
    if (gid >= 2 * NP) return;
    int b = gid >> 13;
    const float* t  = t36 + b * 36;
    const float* xi = x + (size_t)gid * 6;
    float o[6];
    for (int d2 = 0; d2 < 6; ++d2) {
        float s = 0.f;
        for (int c = 0; c < 6; ++c)
            s += xi[c] * (t[c * 6 + d2] + (c == d2 ? 1.f : 0.f));
        o[d2] = s;
    }
    for (int d2 = 0; d2 < 6; ++d2) y[(size_t)gid * 6 + d2] = o[d2];
}

__global__ void k_copycols(const float* __restrict__ src, float* __restrict__ dst,
                           int rows, int cs, int ds, int off) {
    long i = blockIdx.x * (long)blockDim.x + threadIdx.x;
    if (i >= (long)rows * cs) return;
    long r = i / cs, c = i % cs;
    dst[r * ds + off + c] = src[i];
}

__global__ void k_transpose_out(const float* __restrict__ xt, float* __restrict__ out, int C) {
    const int NP = 8192;
    long i = blockIdx.x * (long)blockDim.x + threadIdx.x;
    if (i >= 2L * C * NP) return;
    int n = (int)(i % NP); int ch = (int)((i / NP) % C); int b = (int)(i / ((long)C * NP));
    out[i] = xt[((long)b * NP + n) * C + ch];
}

// ---------------------------------------------------------------------------
// Host side
// ---------------------------------------------------------------------------
static inline float* bump(char*& p, size_t elems) {
    float* r = (float*)p;
    p += ((elems * sizeof(float)) + 255) & ~(size_t)255;
    return r;
}
static inline int* bumpi(char*& p, size_t elems) {
    int* r = (int*)p;
    p += ((elems * sizeof(int)) + 255) & ~(size_t)255;
    return r;
}

extern "C" void kernel_launch(void* const* d_in, const int* in_sizes, int n_in,
                              void* d_out, int out_size, void* d_ws, size_t ws_size,
                              hipStream_t stream) {
    (void)in_sizes; (void)n_in; (void)out_size; (void)ws_size;
    const int NP = 8192, ROWS = 2 * 8192;
    const float* data = (const float*)d_in[0];
    // jax tree-flatten order: dicts alphabetical, 'b' before 'w'.
    auto F = [&](int i) { return (const float*)d_in[i]; };
    const float *dg1l1b=F(1),*dg1l1w=F(2),*dg1l2b=F(3),*dg1l2w=F(4);
    const float *dg2l1b=F(5),*dg2l1w=F(6),*dg2l2b=F(7),*dg2l2w=F(8);
    const float *dg3l1b=F(9),*dg3l1w=F(10),*dg3l2b=F(11),*dg3l2w=F(12);
    const float *edgeb=F(13),*edgew=F(14);
    const float *eg1l1b=F(15),*eg1l1w=F(16),*eg1l2b=F(17),*eg1l2w=F(18);
    const float *eg2l1b=F(19),*eg2l1w=F(20),*eg2l2b=F(21),*eg2l2w=F(22);
    const float *eg3l1b=F(23),*eg3l1w=F(24),*eg3l2b=F(25),*eg3l2w=F(26);
    const float *fib=F(27),*fiw=F(28);
    const float *globb=F(29),*globw=F(30);
    const float *localb=F(31),*localw=F(32);
    const float *outb=F(33),*outw=F(34);
    const float *r1l1b=F(35),*r1l1w=F(36),*r1l2b=F(37),*r1l2w=F(38),*r1scb=F(39),*r1scw=F(40);
    const float *r2l1b=F(41),*r2l1w=F(42),*r2l2b=F(43),*r2l2w=F(44),*r2scb=F(45),*r2scw=F(46);
    const float *c1b=F(47),*c1w=F(48),*c2b=F(49),*c2w=F(50),*c3b=F(51),*c3w=F(52);
    const float *f1b=F(53),*f1w=F(54),*f2b=F(55),*f2w=F(56),*f3b=F(57),*f3w=F(58);

    char* ws = (char*)d_ws;
    float* xT    = bump(ws, (size_t)ROWS * 6);
    float* xstn  = bump(ws, (size_t)ROWS * 6);
    float* h64   = bump(ws, (size_t)ROWS * 64);
    float* h128  = bump(ws, (size_t)ROWS * 128);
    float* h1024 = bump(ws, (size_t)ROWS * 1024);
    float* g     = bump(ws, 2 * 1024);
    float* f1o   = bump(ws, 2 * 512);
    float* f2o   = bump(ws, 2 * 256);
    float* t36   = bump(ws, 2 * 36);
    int* idx_eg1 = bumpi(ws, (size_t)ROWS * 32);
    int* idx_dg1 = bumpi(ws, (size_t)ROWS * 32);
    int* idx_dg2 = bumpi(ws, (size_t)ROWS * 32);
    int* idx_dg3 = bumpi(ws, (size_t)ROWS * 32);
    int* idx_eg2 = bumpi(ws, (size_t)ROWS * 32);
    int* idx_eg3 = bumpi(ws, (size_t)ROWS * 32);
    float* x1    = bump(ws, (size_t)ROWS * 24);
    float* x2    = bump(ws, (size_t)ROWS * 24);
    float* x3    = bump(ws, (size_t)ROWS * 24);
    float* cat72 = bump(ws, (size_t)ROWS * 72);
    float* xl    = bump(ws, (size_t)ROWS * 60);
    float* y1    = bump(ws, (size_t)ROWS * 60);
    float* y2    = bump(ws, (size_t)ROWS * 60);
    float* y3    = bump(ws, (size_t)ROWS * 60);
    float* cat240= bump(ws, (size_t)ROWS * 240);
    float* xg    = bump(ws, (size_t)ROWS * 1024);
    float* xfi   = bump(ws, (size_t)ROWS * 1024);
    float* rh    = bump(ws, (size_t)ROWS * 512);
    float* rsc   = bump(ws, (size_t)ROWS * 512);
    float* r1o   = bump(ws, (size_t)ROWS * 512);
    float* rh2   = bump(ws, (size_t)ROWS * 256);
    float* rsc2  = bump(ws, (size_t)ROWS * 256);
    float* r2o   = bump(ws, (size_t)ROWS * 256);
    float* segt  = bump(ws, (size_t)ROWS * 17);
    float* edgt  = bump(ws, (size_t)ROWS * 2);

    auto gemm = [&](const float* A, const float* W, const float* b, const float* aux,
                    float* o, int M, int K, int N, int epi) {
        dim3 gr((M + 63) / 64, (N + 63) / 64);
        k_gemm<<<gr, 256, 0, stream>>>(A, W, b, aux, o, M, K, N, epi);
    };

    // input transpose (B,6,N) -> (B*N,6)
    k_transpose_in<<<(2 * 6 * NP + 255) / 256, 256, 0, stream>>>(data, xT);

    // one sort of cd(pos) serves eg1 knn + all three dilated selectors
    k_knn_sort<3><<<ROWS, 256, 0, stream>>>(xT, 6,
        idx_eg1, 1, idx_dg1, 200 / 32, idx_dg2, 900 / 32, idx_dg3, 1800 / 32);

    // STN
    gemm(xT,    c1w, c1b, nullptr, h64,   ROWS, 6,    64,   1);
    gemm(h64,   c2w, c2b, nullptr, h128,  ROWS, 64,   128,  1);
    gemm(h128,  c3w, c3b, nullptr, h1024, ROWS, 128,  1024, 1);
    k_colmax<<<dim3(4, 2), 256, 0, stream>>>(h1024, g, 1024);
    gemm(g,   f1w, f1b, nullptr, f1o, 2, 1024, 512, 1);
    gemm(f1o, f2w, f2b, nullptr, f2o, 2, 512,  256, 1);
    gemm(f2o, f3w, f3b, nullptr, t36, 2, 256,  36,  0);
    k_apply_stn<<<(2 * NP + 255) / 256, 256, 0, stream>>>(xT, t36, xstn);

    // edge conv chain
    k_edge_conv<6, 24><<<ROWS / 4, 128, 0, stream>>>(xstn, idx_eg1,
        eg1l1w, eg1l1b, eg1l2w, eg1l2b, x1);
    k_knn_sort<24><<<ROWS, 256, 0, stream>>>(x1, 24,
        idx_eg2, 1, nullptr, 1, nullptr, 1, nullptr, 1);
    k_edge_conv<24, 24><<<ROWS / 4, 128, 0, stream>>>(x1, idx_eg2,
        eg2l1w, eg2l1b, eg2l2w, eg2l2b, x2);
    k_knn_sort<24><<<ROWS, 256, 0, stream>>>(x2, 24,
        idx_eg3, 1, nullptr, 1, nullptr, 1, nullptr, 1);
    k_edge_conv<24, 24><<<ROWS / 4, 128, 0, stream>>>(x2, idx_eg3,
        eg3l1w, eg3l1b, eg3l2w, eg3l2b, x3);

    // local fuse
    k_copycols<<<(ROWS * 24 + 255) / 256, 256, 0, stream>>>(x1, cat72, ROWS, 24, 72, 0);
    k_copycols<<<(ROWS * 24 + 255) / 256, 256, 0, stream>>>(x2, cat72, ROWS, 24, 72, 24);
    k_copycols<<<(ROWS * 24 + 255) / 256, 256, 0, stream>>>(x3, cat72, ROWS, 24, 72, 48);
    gemm(cat72, localw, localb, nullptr, xl, ROWS, 72, 60, 1);

    // dilated edge convs (indices from pos cd sort)
    k_edge_conv<60, 60><<<ROWS / 4, 128, 0, stream>>>(xl, idx_dg1,
        dg1l1w, dg1l1b, dg1l2w, dg1l2b, y1);
    k_edge_conv<60, 60><<<ROWS / 4, 128, 0, stream>>>(y1, idx_dg2,
        dg2l1w, dg2l1b, dg2l2w, dg2l2b, y2);
    k_edge_conv<60, 60><<<ROWS / 4, 128, 0, stream>>>(y2, idx_dg3,
        dg3l1w, dg3l1b, dg3l2w, dg3l2b, y3);

    // global fuse + gated unit
    k_copycols<<<(ROWS * 60 + 255) / 256, 256, 0, stream>>>(xl, cat240, ROWS, 60, 240, 0);
    k_copycols<<<(ROWS * 60 + 255) / 256, 256, 0, stream>>>(y1, cat240, ROWS, 60, 240, 60);
    k_copycols<<<(ROWS * 60 + 255) / 256, 256, 0, stream>>>(y2, cat240, ROWS, 60, 240, 120);
    k_copycols<<<(ROWS * 60 + 255) / 256, 256, 0, stream>>>(y3, cat240, ROWS, 60, 240, 180);
    gemm(cat240, globw, globb, nullptr, xg, ROWS, 240, 1024, 1);
    gemm(xg, fiw, fib, xg, xfi, ROWS, 1024, 1024, 2);        // x * sigmoid(fi(x))

    // residual blocks
    gemm(xfi, r1l1w, r1l1b, nullptr, rh,  ROWS, 1024, 512, 1);
    gemm(xfi, r1scw, r1scb, nullptr, rsc, ROWS, 1024, 512, 0);
    gemm(rh,  r1l2w, r1l2b, rsc,     r1o, ROWS, 512,  512, 3);
    gemm(r1o, r2l1w, r2l1b, nullptr, rh2,  ROWS, 512, 256, 1);
    gemm(r1o, r2scw, r2scb, nullptr, rsc2, ROWS, 512, 256, 0);
    gemm(rh2, r2l2w, r2l2b, rsc2,    r2o,  ROWS, 256, 256, 3);

    // heads
    gemm(r2o, outw,  outb,  nullptr, segt, ROWS, 256, 17, 0);
    gemm(r2o, edgew, edgeb, nullptr, edgt, ROWS, 256, 2,  0);

    float* out = (float*)d_out;
    k_transpose_out<<<(2 * 17 * NP + 255) / 256, 256, 0, stream>>>(segt, out, 17);
    k_transpose_out<<<(2 * 2 * NP + 255) / 256, 256, 0, stream>>>(edgt, out + 2L * 17 * NP, 2);
}
